// SwinEncoder_16475494547493
// MI455X (gfx1250) — matrix operand order repaired and checked by reading, not compiled, not run on hardware
//
#include <hip/hip_runtime.h>
#include <math.h>

// ============================================================================
// CDNA5 (gfx1250) Swin encoder forward.  wave32, WMMA bf16 16x16x32 path.
// Activations fp32 in HBM, converted to bf16 in-flight into LDS for all GEMM
// work (QKV / attn / proj / MLP / merge / head) -> v_wmma_f32_16x16x32_bf16.
// All GEMM shapes in this net are M%64==0, N%64==0, K%32==0 -> no tail code.
// ============================================================================

typedef __attribute__((ext_vector_type(16))) __bf16 v16bf;
typedef __attribute__((ext_vector_type(8)))  float  v8f;

union AFrag { v16bf v; unsigned int u[8]; };

__device__ __forceinline__ unsigned short f2bf(float f) {
  union { float f; unsigned int u; } a; a.f = f;
  unsigned int r = a.u + 0x7fffu + ((a.u >> 16) & 1u);   // round-to-nearest-even
  return (unsigned short)(r >> 16);
}
__device__ __forceinline__ unsigned int pack2(float lo, float hi) {
  return (unsigned int)f2bf(lo) | ((unsigned int)f2bf(hi) << 16);
}

// ---------------------------------------------------------------------------
// Patch embed: conv 4x4 stride 4, 3->64, NCHW input -> (B, 56*56, 64) output
// ---------------------------------------------------------------------------
__global__ __launch_bounds__(256) void patch_embed_k(
    const float* __restrict__ x, const float* __restrict__ w,
    const float* __restrict__ bias, float* __restrict__ out, int B)
{
  size_t idx = (size_t)blockIdx.x * 256 + threadIdx.x;
  size_t total = (size_t)B * 3136 * 64;
  if (idx >= total) return;
  int c  = (int)(idx & 63);
  size_t t = idx >> 6;
  int xy = (int)(t % 3136);
  int b  = (int)(t / 3136);
  int oy = xy / 56, ox = xy % 56;
  float acc = bias[c];
  const float* xb = x + (size_t)b * 3 * 224 * 224;
  const float* wc = w + (size_t)c * 48;          // [c][3][4][4]
  #pragma unroll
  for (int ci = 0; ci < 3; ci++)
    #pragma unroll
    for (int ky = 0; ky < 4; ky++)
      #pragma unroll
      for (int kx = 0; kx < 4; kx++)
        acc += xb[((size_t)ci * 224 + oy * 4 + ky) * 224 + ox * 4 + kx]
             * wc[(ci * 4 + ky) * 4 + kx];
  out[idx] = acc;
}

// ---------------------------------------------------------------------------
// LayerNorm over last dim C; one wave32 per row.
// ---------------------------------------------------------------------------
__global__ __launch_bounds__(256) void layernorm_k(
    const float* __restrict__ in, const float* __restrict__ g,
    const float* __restrict__ b, float* __restrict__ out, int rows, int C)
{
  int wv = threadIdx.x >> 5, lane = threadIdx.x & 31;
  int row = blockIdx.x * 8 + wv;
  if (row >= rows) return;
  const float* xr = in + (size_t)row * C;
  float s = 0.f, s2 = 0.f;
  for (int c = lane; c < C; c += 32) { float v = xr[c]; s += v; s2 += v * v; }
  #pragma unroll
  for (int off = 16; off; off >>= 1) {
    s  += __shfl_xor(s,  off, 32);
    s2 += __shfl_xor(s2, off, 32);
  }
  float m   = s / C;
  float inv = rsqrtf(fmaxf(s2 / C - m * m, 0.f) + 1e-5f);
  float* orow = out + (size_t)row * C;
  for (int c = lane; c < C; c += 32) orow[c] = (xr[c] - m) * inv * g[c] + b[c];
}

// ---------------------------------------------------------------------------
// Generic GEMM:  out[M,N] = epilogue(A[M,K] @ W[K,N])
//   mode 0: + bias (if non-null)
//   mode 1: + bias, exact GELU
//   mode 2: + bias, + resid[M,N]
// 64x64 tile / 256 threads (8 waves, 2 accum tiles each), BK=32, bf16 WMMA.
// PRECONDITION: M%64==0, N%64==0, K%32==0 (true for every call in this net).
// ---------------------------------------------------------------------------
__global__ __launch_bounds__(256) void gemm_bf16_wmma(
    const float* __restrict__ A, const float* __restrict__ Wp,
    const float* __restrict__ bias, const float* __restrict__ resid,
    float* __restrict__ out, int M, int N, int K, int mode)
{
  __shared__ unsigned short As[64 * 40];   // A tile, row-major [m][k], stride 40
  __shared__ unsigned short Bs[64 * 40];   // W tile transposed [n][k], stride 40

  int tid  = threadIdx.x;
  int m0   = blockIdx.y * 64, n0 = blockIdx.x * 64;
  int wave = tid >> 5, lane = tid & 31, half = lane >> 4, l16 = lane & 15;
  int tm   = wave >> 1;             // 0..3 : M sub-tile
  int tn0  = (wave & 1) * 2;        // 0 or 2 : first of two N sub-tiles

  // staging assignments (no bounds checks needed)
  int ar = tid >> 2, acg = (tid & 3) * 8;        // A: row 0..63, 8 consecutive k
  int bn = tid & 63, bk8 = (tid >> 6) * 8;       // B: col n, 8 consecutive k
  const float* Arow = A + (size_t)(m0 + ar) * K + acg;
  const float* Bcol = Wp + (size_t)bk8 * N + n0 + bn;

  v8f acc0 = {}; v8f acc1 = {};

  for (int k0 = 0; k0 < K; k0 += 32) {
    // prefetch next K-slab into L2/L0 while this one is consumed
    if (k0 + 32 < K) {
      __builtin_prefetch(Arow + k0 + 32, 0, 1);
      __builtin_prefetch(Bcol + (size_t)(k0 + 32) * N, 0, 1);
    }
    { // stage A 64x32: two b128 loads -> one b128 LDS store (8 bf16)
      float4 f0 = *(const float4*)(Arow + k0);
      float4 f1 = *(const float4*)(Arow + k0 + 4);
      uint4 q;
      q.x = pack2(f0.x, f0.y); q.y = pack2(f0.z, f0.w);
      q.z = pack2(f1.x, f1.y); q.w = pack2(f1.z, f1.w);
      *(uint4*)&As[ar * 40 + acg] = q;
    }
    { // stage W 32x64 transposed: 8 coalesced b32 loads -> one b128 LDS store
      const float* p = Bcol + (size_t)k0 * N;
      float v0 = p[0];
      float v1 = p[(size_t)1 * N];
      float v2 = p[(size_t)2 * N];
      float v3 = p[(size_t)3 * N];
      float v4 = p[(size_t)4 * N];
      float v5 = p[(size_t)5 * N];
      float v6 = p[(size_t)6 * N];
      float v7 = p[(size_t)7 * N];
      uint4 q;
      q.x = pack2(v0, v1); q.y = pack2(v2, v3);
      q.z = pack2(v4, v5); q.w = pack2(v6, v7);
      *(uint4*)&Bs[bn * 40 + bk8] = q;
    }
    __syncthreads();

    AFrag a;   // 16-bit A 16x32 layout (ISA 7.12.2)
    #pragma unroll
    for (int j = 0; j < 8; j++) {
      int kk = ((j < 4) ? 0 : 16) + 2 * (j & 3) + half * 8;
      a.u[j] = *(const unsigned int*)&As[(tm * 16 + l16) * 40 + kk];
    }
    AFrag b0, b1;  // 16-bit B 32x16 layout: lane=n, same K order as A
    #pragma unroll
    for (int j = 0; j < 8; j++) {
      int kk = ((j < 4) ? 0 : 16) + 2 * (j & 3) + half * 8;
      b0.u[j] = *(const unsigned int*)&Bs[(tn0 * 16 + l16) * 40 + kk];
      b1.u[j] = *(const unsigned int*)&Bs[((tn0 + 1) * 16 + l16) * 40 + kk];
    }
    acc0 = __builtin_amdgcn_wmma_f32_16x16x32_bf16(false, a.v, false, b0.v,
                                                   (short)0, acc0, false, false);
    acc1 = __builtin_amdgcn_wmma_f32_16x16x32_bf16(false, a.v, false, b1.v,
                                                   (short)0, acc1, false, false);
    __syncthreads();
  }

  // epilogue; C/D layout: VGPR r -> M = r + 8*half, lane%16 -> N
  #pragma unroll
  for (int ht = 0; ht < 2; ht++) {
    v8f acc = ht ? acc1 : acc0;
    int tn = tn0 + ht;
    int gn = n0 + tn * 16 + l16;
    float bv = bias ? bias[gn] : 0.f;
    #pragma unroll
    for (int r = 0; r < 8; r++) {
      int gm = m0 + tm * 16 + r + half * 8;
      float v = acc[r] + bv;
      if (mode == 1)      v = 0.5f * v * (1.f + erff(v * 0.70710678f));
      else if (mode == 2) v += resid[(size_t)gm * N + gn];
      out[(size_t)gm * N + gn] = v;
    }
  }
}

// ---------------------------------------------------------------------------
// Window attention: one workgroup (4 waves) per (image, window, head).
// 49 tokens padded to 64; hd = 32 everywhere.  Cyclic shift folded into
// gather (qkv read) and scatter (output write).
// ---------------------------------------------------------------------------
__device__ __forceinline__ long long tok_row(int b, int t, int wy, int wx,
                                             int H, int W, int shift)
{
  int iy = t / 7, ix = t % 7;
  int y = wy * 7 + iy, x = wx * 7 + ix;
  if (shift) {
    y += shift; if (y >= H) y -= H;
    x += shift; if (x >= W) x -= W;
  }
  return (long long)b * H * W + (long long)y * W + x;
}

__global__ __launch_bounds__(128) void win_attn_k(
    const float* __restrict__ qkv, float* __restrict__ out,
    int H, int W, int C, int heads, int shift)
{
  __shared__ unsigned short qs[64 * 40];   // q (scaled), [t][d]
  __shared__ unsigned short ks[64 * 40];   // k,          [t][d]  (acts as B^T)
  __shared__ unsigned short vt[32 * 72];   // v transposed [d][t]
  __shared__ unsigned short ps[64 * 72];   // softmax probs bf16 [t][t']
  __shared__ float          ss[64 * 68];   // scores f32

  int nwx = W / 7, nW = (H / 7) * nwx;
  int id   = blockIdx.x;
  int head = id % heads;
  int win  = (id / heads) % nW;
  int b    = id / (heads * nW);
  int wy = win / nwx, wx = win % nwx;
  int C3 = 3 * C;
  const float scale = 0.1767766953f;       // 32^-0.5

  int tid = threadIdx.x;
  // ---- stage q/k/v into LDS (fp32 -> bf16), zero-pad rows 49..63 ----
  // 64 tokens x 8 groups of 4 channels = 512 work items
  for (int i = tid; i < 64 * 8; i += 128) {
    int t = i >> 3, d = (i & 7) * 4;
    float4 q4 = make_float4(0.f, 0.f, 0.f, 0.f);
    float4 k4 = q4, v4 = q4;
    if (t < 49) {
      long long r = tok_row(b, t, wy, wx, H, W, shift);
      const float* p = qkv + (size_t)r * C3 + head * 32 + d;
      q4 = *(const float4*)p;
      k4 = *(const float4*)(p + C);
      v4 = *(const float4*)(p + 2 * C);
      q4.x *= scale; q4.y *= scale; q4.z *= scale; q4.w *= scale;
    }
    uint2 qq; qq.x = pack2(q4.x, q4.y); qq.y = pack2(q4.z, q4.w);
    uint2 kk; kk.x = pack2(k4.x, k4.y); kk.y = pack2(k4.z, k4.w);
    *(uint2*)&qs[t * 40 + d] = qq;
    *(uint2*)&ks[t * 40 + d] = kk;
    vt[(d + 0) * 72 + t] = f2bf(v4.x);
    vt[(d + 1) * 72 + t] = f2bf(v4.y);
    vt[(d + 2) * 72 + t] = f2bf(v4.z);
    vt[(d + 3) * 72 + t] = f2bf(v4.w);
  }
  __syncthreads();

  int wave = tid >> 5, lane = tid & 31, half = lane >> 4, l16 = lane & 15;

  // ---- S = q @ k^T  (4x4 tiles of 16x16, K=32 in one WMMA) ----
  {
    int tm = wave;
    AFrag a;
    #pragma unroll
    for (int j = 0; j < 8; j++) {
      int kk = ((j < 4) ? 0 : 16) + 2 * (j & 3) + half * 8;
      a.u[j] = *(const unsigned int*)&qs[(tm * 16 + l16) * 40 + kk];
    }
    for (int tn = 0; tn < 4; tn++) {
      AFrag bf_;
      #pragma unroll
      for (int j = 0; j < 8; j++) {
        int kk = ((j < 4) ? 0 : 16) + 2 * (j & 3) + half * 8;
        bf_.u[j] = *(const unsigned int*)&ks[(tn * 16 + l16) * 40 + kk];
      }
      v8f acc = {};
      acc = __builtin_amdgcn_wmma_f32_16x16x32_bf16(false, a.v, false, bf_.v,
                                                    (short)0, acc, false, false);
      #pragma unroll
      for (int r = 0; r < 8; r++)
        ss[(tm * 16 + r + 8 * half) * 68 + tn * 16 + l16] = acc[r];
    }
  }
  __syncthreads();

  // ---- row softmax over 49 valid cols; write bf16 P, zero padding ----
  if (tid < 64) {
    int r = tid;
    if (r < 49) {
      float mx = -1e30f;
      for (int c = 0; c < 49; c++) mx = fmaxf(mx, ss[r * 68 + c]);
      float sum = 0.f;
      for (int c = 0; c < 49; c++) {
        float e = __expf(ss[r * 68 + c] - mx);
        ss[r * 68 + c] = e; sum += e;
      }
      float inv = 1.f / sum;
      for (int c = 0; c < 49; c++) ps[r * 72 + c] = f2bf(ss[r * 68 + c] * inv);
      for (int c = 49; c < 64; c++) ps[r * 72 + c] = 0;
    } else {
      for (int c = 0; c < 64; c++) ps[r * 72 + c] = 0;
    }
  }
  __syncthreads();

  // ---- O = P @ V   (4 M-tiles x 2 N-tiles, K=64 in two WMMA steps) ----
  {
    int tm = wave;
    for (int tn = 0; tn < 2; tn++) {
      v8f acc = {};
      #pragma unroll
      for (int kst = 0; kst < 2; kst++) {
        AFrag a;
        #pragma unroll
        for (int j = 0; j < 8; j++) {
          int kk = kst * 32 + ((j < 4) ? 0 : 16) + 2 * (j & 3) + half * 8;
          a.u[j] = *(const unsigned int*)&ps[(tm * 16 + l16) * 72 + kk];
        }
        AFrag bf_;
        #pragma unroll
        for (int j = 0; j < 8; j++) {
          int kk = kst * 32 + ((j < 4) ? 0 : 16) + 2 * (j & 3) + half * 8;
          bf_.u[j] = *(const unsigned int*)&vt[(tn * 16 + l16) * 72 + kk];
        }
        acc = __builtin_amdgcn_wmma_f32_16x16x32_bf16(false, a.v, false, bf_.v,
                                                      (short)0, acc, false, false);
      }
      #pragma unroll
      for (int r = 0; r < 8; r++) {
        int m = tm * 16 + r + 8 * half;
        if (m < 49) {
          long long rr = tok_row(b, m, wy, wx, H, W, shift);
          out[(size_t)rr * C + head * 32 + tn * 16 + l16] = acc[r];
        }
      }
    }
  }
}

// ---------------------------------------------------------------------------
// Patch merging gather: (B,H,W,C) -> (B, H/2*W/2, 4C), 4C = [sy][sx][C]
// ---------------------------------------------------------------------------
__global__ __launch_bounds__(256) void merge_gather_k(
    const float* __restrict__ x, float* __restrict__ out, int B, int H, int W, int C)
{
  int H2 = H / 2, W2 = W / 2;
  size_t total = (size_t)B * H2 * W2 * 4 * C;
  size_t idx = (size_t)blockIdx.x * 256 + threadIdx.x;
  if (idx >= total) return;
  int C4 = 4 * C;
  int c4 = (int)(idx % C4);
  size_t t = idx / C4;
  int tok = (int)(t % (H2 * W2));
  int b   = (int)(t / (H2 * W2));
  int grp = c4 / C, c = c4 % C;
  int sy = grp >> 1, sx = grp & 1;
  int y2 = tok / W2, x2 = tok % W2;
  out[idx] = x[((size_t)b * H * W + (size_t)(2 * y2 + sy) * W + (2 * x2 + sx)) * C + c];
}

// ===========================================================================
// Host orchestration
// ===========================================================================
extern "C" void kernel_launch(void* const* d_in, const int* in_sizes, int n_in,
                              void* d_out, int out_size, void* d_ws, size_t ws_size,
                              hipStream_t stream)
{
  const int B = 32;

  // jax pytree flattening order (dicts sorted by key):
  // 0 norm_b, 1 norm_g, 2 patch_b, 3 patch_w, 4 proj_b, 5 proj_w, then stages.
  // per block (sorted): fc1_b, fc1_w, fc2_b, fc2_w, n1b, n1g, n2b, n2g,
  //                     proj_b, proj_w, qkv_b, qkv_w
  // per merge (sorted): nb, ng, red_w
  const float* norm_b  = (const float*)d_in[0];
  const float* norm_g  = (const float*)d_in[1];
  const float* patch_b = (const float*)d_in[2];
  const float* patch_w = (const float*)d_in[3];
  const float* proj_b  = (const float*)d_in[4];
  const float* proj_w  = (const float*)d_in[5];

  const float* blk[4][12];      // s0b0, s1b0, s2b0, s2b1
  const float* mrg[2][3];       // merge0 (nb,ng,red_w), merge1
  int i = 6;
  for (int j = 0; j < 12; j++) blk[0][j] = (const float*)d_in[i++];   // 6..17
  for (int j = 0; j < 3;  j++) mrg[0][j] = (const float*)d_in[i++];   // 18..20
  for (int j = 0; j < 12; j++) blk[1][j] = (const float*)d_in[i++];   // 21..32
  for (int j = 0; j < 3;  j++) mrg[1][j] = (const float*)d_in[i++];   // 33..35
  for (int j = 0; j < 12; j++) blk[2][j] = (const float*)d_in[i++];   // 36..47
  for (int j = 0; j < 12; j++) blk[3][j] = (const float*)d_in[i++];   // 48..59

  // locate the image tensor robustly by its unique size
  int xi = n_in - 1;
  for (int k = 0; k < n_in; k++)
    if (in_sizes[k] == 32 * 3 * 224 * 224) { xi = k; break; }
  const float* ximg = (const float*)d_in[xi];

  // workspace layout (floats); requires ~180 MB
  const size_t SB = 6423552;                 // >= 32*3136*64, padded
  float* XA  = (float*)d_ws;
  float* XB  = XA + SB;
  float* Hb  = XB + SB;
  float* AO  = Hb + SB;                      // attn-out / merge-LN scratch
  float* BIG = AO + SB;                      // qkv / mlp-hidden / merge gather (19.3M)

  auto gemm = [&](const float* A, const float* Wp, const float* bia,
                  const float* res, float* o, int M, int N, int K, int mode) {
    dim3 g((N + 63) / 64, (M + 63) / 64);
    gemm_bf16_wmma<<<g, dim3(256), 0, stream>>>(A, Wp, bia, res, o, M, N, K, mode);
  };
  auto ln = [&](const float* in, const float* g_, const float* b_, float* o,
                int rows, int C) {
    layernorm_k<<<dim3((rows + 7) / 8), dim3(256), 0, stream>>>(in, g_, b_, o, rows, C);
  };

  auto run_block = [&](const float* const* p, const float* xin, float* xout,
                       int Hh, int Ww, int C, int heads, int shift) {
    int M = B * Hh * Ww, hid = 3 * C;
    // p: 0 fc1_b 1 fc1_w 2 fc2_b 3 fc2_w 4 n1b 5 n1g 6 n2b 7 n2g
    //    8 proj_b 9 proj_w 10 qkv_b 11 qkv_w
    ln(xin, p[5], p[4], Hb, M, C);
    gemm(Hb, p[11], p[10], nullptr, BIG, M, 3 * C, C, 0);              // QKV
    int nblk = B * (Hh / 7) * (Ww / 7) * heads;
    win_attn_k<<<dim3(nblk), dim3(128), 0, stream>>>(BIG, AO, Hh, Ww, C, heads, shift);
    gemm(AO, p[9], p[8], xin, xout, M, C, C, 2);                       // proj + resid
    ln(xout, p[7], p[6], Hb, M, C);
    gemm(Hb, p[1], p[0], nullptr, BIG, M, hid, C, 1);                  // fc1 + GELU
    gemm(BIG, p[3], p[2], xout, xout, M, C, hid, 2);                   // fc2 + resid
  };

  auto run_merge = [&](const float* const* m, const float* xin, float* xout,
                       int Hh, int Ww, int C) {
    int H2 = Hh / 2, W2 = Ww / 2, M2 = B * H2 * W2, C4 = 4 * C;
    size_t total = (size_t)M2 * C4;
    merge_gather_k<<<dim3((unsigned)((total + 255) / 256)), dim3(256), 0, stream>>>(
        xin, BIG, B, Hh, Ww, C);
    ln(BIG, m[1], m[0], AO, M2, C4);                                   // ng, nb
    gemm(AO, m[2], nullptr, nullptr, xout, M2, 2 * C, C4, 0);          // red_w
  };

  // ---- pipeline ----
  {
    size_t total = (size_t)B * 3136 * 64;
    patch_embed_k<<<dim3((unsigned)((total + 255) / 256)), dim3(256), 0, stream>>>(
        ximg, patch_w, patch_b, XA, B);
  }
  run_block(blk[0], XA, XB, 56, 56, 64, 2, 0);        // stage 0
  run_merge(mrg[0], XB, XA, 56, 56, 64);              // -> 28x28, C=128
  run_block(blk[1], XA, XB, 28, 28, 128, 4, 0);       // stage 1
  run_merge(mrg[1], XB, XA, 28, 28, 128);             // -> 14x14, C=256
  run_block(blk[2], XA, XB, 14, 14, 256, 8, 0);       // stage 2 block 0
  run_block(blk[3], XB, XA, 14, 14, 256, 8, 3);       // stage 2 block 1 (shift)

  // final norm + head -> d_out (32*196*256 fp32)
  {
    int M = B * 196;
    ln(XA, norm_g, norm_b, Hb, M, 256);
    gemm(Hb, proj_w, proj_b, nullptr, (float*)d_out, M, 256, 256, 0);
  }
}
